// KPConv_70059506532627
// MI455X (gfx1250) — matrix operand order repaired and checked by reading, not compiled, not run on hardware
//
#include <hip/hip_runtime.h>

typedef __attribute__((ext_vector_type(16))) _Float16 v16h;
typedef __attribute__((ext_vector_type(8)))  float    v8f;
typedef __attribute__((ext_vector_type(2)))  _Float16 h2v;

union FragH {
    v16h   v;
    float4 f4[2];
    _Float16 h[16];
};

#define H_NB  32    // neighbors per query
#define KPTS  15    // kernel points
#define CIN   64
#define COUT  64
#define CKDIM 1024  // flattened (c,k): c*16 + k, k padded to 16; = 32 k-steps of 32

__device__ __forceinline__ void store_h2(_Float16* p, float a, float b) {
    h2v v;
    v[0] = (_Float16)a;
    v[1] = (_Float16)b;
    *(h2v*)p = v;
}

// ---------------------------------------------------------------------------
// Prep: weights (15,64,64) f32 -> f16, reindexed by ck = c*16 + k (k=15 -> 0)
// and pre-swizzled into per-lane WMMA B-fragment order: for k-step s (0..31),
// cout-tile t (0..3), lane (0..31): 32 contiguous bytes = 8 half2; half2 v
// holds W[ck], W[ck+1] at column d, ck = 32*s + kloB + 2*v,
// kloB = (lane<16)?0:16, d = 16*t + (lane&15).
// ---------------------------------------------------------------------------
__global__ void kpconv_prep_weights(const float* __restrict__ weights,
                                    _Float16* __restrict__ wB)
{
    int gid = blockIdx.x * blockDim.x + threadIdx.x;   // one half2 per thread
    if (gid >= 32 * 4 * 32 * 8) return;
    int v    = gid & 7;
    int lane = (gid >> 3) & 31;
    int t    = (gid >> 8) & 3;
    int s    = gid >> 10;                              // 0..31
    int kloB = (lane < 16) ? 0 : 16;
    int ck   = 32 * s + kloB + 2 * v;                  // even, <= 1022
    int d    = 16 * t + (lane & 15);
    int c    = ck >> 4;
    int k0   = ck & 15;                                // even -> k0 <= 14 valid
    float w0 = weights[(k0 * CIN + c) * COUT + d];
    float w1 = (k0 + 1 < KPTS) ? weights[((k0 + 1) * CIN + c) * COUT + d] : 0.0f;
    wB[2 * gid + 0] = (_Float16)w0;
    wB[2 * gid + 1] = (_Float16)w1;
}

// ---------------------------------------------------------------------------
// Main kernel: one workgroup (4 wave32) per 16 query points.
// ---------------------------------------------------------------------------
__global__ __launch_bounds__(128)
void kpconv_main(const float* __restrict__ q_points,
                 const float* __restrict__ s_points,
                 const float* __restrict__ s_feats,
                 const int*   __restrict__ nb_idx,
                 const float* __restrict__ kernel_pts,
                 const _Float16* __restrict__ wB,
                 float* __restrict__ out,
                 int M, int N)
{
    __shared__ __align__(16) _Float16 s_infl[16][16][H_NB];    // [mi][k][h]  16 KB
    __shared__ __align__(16) _Float16 s_featT[16][CIN][H_NB];  // [mi][c][h]  64 KB
    __shared__ __align__(16) _Float16 s_wgt[16][CKDIM];        // [mi][c*16+k] 32 KB
    __shared__ float s_kp[KPTS][3];
    __shared__ float s_q[16][3];
    __shared__ int   s_nnum[16];

    const int tid   = threadIdx.x;
    const int lane  = tid & 31;
    const int wave  = tid >> 5;
    const int mBase = blockIdx.x * 16;

    if (tid < KPTS * 3) ((float*)s_kp)[tid] = kernel_pts[tid];
    if (tid < 48) {
        int g = mBase * 3 + tid;
        ((float*)s_q)[tid] = (g < M * 3) ? q_points[g] : 0.0f;
    }
    if (tid < 16) s_nnum[tid] = 0;
    __syncthreads();

    // ---------------- Phase 0: gather + influence, h-pair per task --------
    #pragma unroll
    for (int i = 0; i < 2; ++i) {
        int task = tid + 128 * i;          // 0..255 = 16 mi x 16 h-pairs
        int mi = task >> 4;
        int h0 = (task & 15) * 2;          // even h
        int m  = mBase + mi;

        const int* ip = nb_idx + (size_t)m * H_NB + h0;
        int idx0 = (m < M) ? ip[0] : N;
        int idx1 = (m < M) ? ip[1] : N;
        bool r0 = (idx0 >= 0) && (idx0 < N);
        bool r1 = (idx1 >= 0) && (idx1 < N);
        int sa = r0 ? idx0 : 0;
        int sb = r1 ? idx1 : 0;

        float qx = s_q[mi][0], qy = s_q[mi][1], qz = s_q[mi][2];
        float p0x = s_points[sa * 3 + 0], p0y = s_points[sa * 3 + 1], p0z = s_points[sa * 3 + 2];
        float p1x = s_points[sb * 3 + 0], p1y = s_points[sb * 3 + 1], p1z = s_points[sb * 3 + 2];
        if (!r0) { p0x = 1e6f; p0y = 1e6f; p0z = 1e6f; }
        if (!r1) { p1x = 1e6f; p1y = 1e6f; p1z = 1e6f; }
        float n0x = p0x - qx, n0y = p0y - qy, n0z = p0z - qz;
        float n1x = p1x - qx, n1y = p1y - qy, n1z = p1z - qz;

        // features: gather two rows, pack (h0,h1) pairs -> single b32 stores
        float fs0 = 0.0f, fs1 = 0.0f;
        {
            const float4* fa = (const float4*)(s_feats + (size_t)sa * CIN);
            const float4* fb = (const float4*)(s_feats + (size_t)sb * CIN);
            #pragma unroll
            for (int c4 = 0; c4 < CIN / 4; ++c4) {
                float4 a = fa[c4];
                float4 b = fb[c4];
                if (!r0) { a.x = 0.f; a.y = 0.f; a.z = 0.f; a.w = 0.f; }
                if (!r1) { b.x = 0.f; b.y = 0.f; b.z = 0.f; b.w = 0.f; }
                fs0 += a.x + a.y + a.z + a.w;
                fs1 += b.x + b.y + b.z + b.w;
                int c = 4 * c4;
                store_h2(&s_featT[mi][c + 0][h0], a.x, b.x);
                store_h2(&s_featT[mi][c + 1][h0], a.y, b.y);
                store_h2(&s_featT[mi][c + 2][h0], a.z, b.z);
                store_h2(&s_featT[mi][c + 3][h0], a.w, b.w);
            }
        }
        int val = (fs0 > 0.0f ? 1 : 0) + (fs1 > 0.0f ? 1 : 0);
        if (val) atomicAdd(&s_nnum[mi], val);

        // influence[k][h] = max(1 - |p - kp_k|, 0)   (SIGMA = 1)
        #pragma unroll
        for (int k = 0; k < KPTS; ++k) {
            float kx = s_kp[k][0], ky = s_kp[k][1], kz = s_kp[k][2];
            float dx0 = n0x - kx, dy0 = n0y - ky, dz0 = n0z - kz;
            float dx1 = n1x - kx, dy1 = n1y - ky, dz1 = n1z - kz;
            float d0 = sqrtf(dx0 * dx0 + dy0 * dy0 + dz0 * dz0);
            float d1 = sqrtf(dx1 * dx1 + dy1 * dy1 + dz1 * dz1);
            store_h2(&s_infl[mi][k][h0], fmaxf(1.0f - d0, 0.0f), fmaxf(1.0f - d1, 0.0f));
        }
        store_h2(&s_infl[mi][15][h0], 0.0f, 0.0f);   // pad row
    }
    __syncthreads();

    // ---------------- Phase 1: weighted = infl^T @ feats ------------------
    // A: 16x32 (rows=k, K=h); B: 32x16 (K=h, cols=c); one WMMA per c-tile.
    // D stored f16 at s_wgt[mi][c*16 + k]; adjacent D regs (k,k+1) -> b32.
    {
        const int kRow = lane & 15;
        const int kloA = (lane < 16) ? 0 : 8;
        const int kloB = (lane < 16) ? 0 : 16;
        const int kHi  = (lane < 16) ? 0 : 8;
        for (int j = 0; j < 4; ++j) {
            int mi = wave * 4 + j;
            FragH a;
            a.f4[0] = *(const float4*)&s_infl[mi][kRow][kloA];
            a.f4[1] = *(const float4*)&s_infl[mi][kRow][kloA + 16];
            #pragma unroll
            for (int t = 0; t < 4; ++t) {
                int c = 16 * t + (lane & 15);
                FragH b;
                b.f4[0] = *(const float4*)&s_featT[mi][c][kloA];
                b.f4[1] = *(const float4*)&s_featT[mi][c][kloA + 16];
                v8f acc = {};
                acc = __builtin_amdgcn_wmma_f32_16x16x32_f16(
                        false, a.v, false, b.v, (short)0, acc, false, false);
                // D: vgpr r -> row k = r + kHi, col c; pack (r, r+1)
                #pragma unroll
                for (int r = 0; r < 8; r += 2) {
                    int k0 = r + kHi;
                    store_h2(&s_wgt[mi][c * 16 + k0], acc[r], acc[r + 1]);
                }
            }
        }
    }
    __syncthreads();

    // ---------------- Phase 2: out = weighted(16x1024) @ W(1024x64) -------
    {
        const int d0   = 16 * wave;           // this wave's Cout tile
        const int col  = lane & 15;           // A row (=m) and B/D column
        const int kloA = (lane < 16) ? 0 : 16;
        const float4* ws4 = (const float4*)wB;

        v8f acc = {};
        for (int s = 0; s < 32; ++s) {
            int base = 32 * s;
            FragH a, b;
            a.f4[0] = *(const float4*)&s_wgt[col][base + kloA];
            a.f4[1] = *(const float4*)&s_wgt[col][base + kloA + 8];
            size_t i4 = ((size_t)(s * 4 + wave) * 32 + lane) * 2;
            b.f4[0] = ws4[i4];
            b.f4[1] = ws4[i4 + 1];
            acc = __builtin_amdgcn_wmma_f32_16x16x32_f16(
                    false, a.v, false, b.v, (short)0, acc, false, false);
        }

        #pragma unroll
        for (int r = 0; r < 8; ++r) {
            int mLoc = r + ((lane < 16) ? 0 : 8);
            int m = mBase + mLoc;
            if (m < M) {
                int nn = s_nnum[mLoc];
                float inv = 1.0f / (float)(nn < 1 ? 1 : nn);
                out[(size_t)m * COUT + d0 + col] = acc[r] * inv;
            }
        }
    }
}

// ---------------------------------------------------------------------------
extern "C" void kernel_launch(void* const* d_in, const int* in_sizes, int n_in,
                              void* d_out, int out_size, void* d_ws, size_t ws_size,
                              hipStream_t stream)
{
    const float* q_points   = (const float*)d_in[0];
    const float* s_points   = (const float*)d_in[1];
    const float* s_feats    = (const float*)d_in[2];
    const int*   nb_idx     = (const int*)d_in[3];
    const float* kernel_pts = (const float*)d_in[4];
    const float* weights    = (const float*)d_in[5];
    float* out = (float*)d_out;

    int M = in_sizes[0] / 3;
    int N = in_sizes[1] / 3;

    _Float16* wB = (_Float16*)d_ws;   // 1024*64 f16 = 128 KB swizzled weights

    kpconv_prep_weights<<<(32 * 4 * 32 * 8 + 255) / 256, 256, 0, stream>>>(weights, wB);

    int blocks = (M + 15) / 16;
    kpconv_main<<<blocks, 128, 0, stream>>>(q_points, s_points, s_feats, nb_idx,
                                            kernel_pts, wB, out, M, N);
}
